// NeuroCore_27144193311197
// MI455X (gfx1250) — matrix operand order repaired and checked by reading, not compiled, not run on hardware
//
#include <hip/hip_runtime.h>
#include <hip/hip_bf16.h>

typedef __attribute__((ext_vector_type(16))) _Float16 v16h;
typedef __attribute__((ext_vector_type(8)))  _Float16 v8h;
typedef __attribute__((ext_vector_type(4)))  _Float16 v4h;
typedef __attribute__((ext_vector_type(8)))  float    v8f;

#define NVARS   100000
#define NLITS   200000
#define NCLS    400000
#define HID     128
#define NEDGES  600000
#define NROUND  4

// ---------------------------------------------------------------------------
// Weight prep: fp32 row-major W[K][N] -> f16 column-major Wt[N][K]
// (one contiguous 32B line per lane per B-fragment afterwards)
// ---------------------------------------------------------------------------
__global__ __launch_bounds__(256) void wprep_kernel(
    const float* __restrict__ W, _Float16* __restrict__ Wt, int K, int N)
{
  int i = blockIdx.x * 256 + threadIdx.x;
  if (i >= K * N) return;
  int n = i / K, k = i - n * K;
  Wt[i] = (_Float16)W[(size_t)k * N + n];
}

// ---------------------------------------------------------------------------
// Cooperative copy of a 128-column f16 weight matrix (col-major, stride kd)
// into LDS with stride kd+8 (16B pad -> fragment reads stagger 4 banks/lane,
// conflict-free across the 64 LDS banks). 16B chunks, fully coalesced.
// ---------------------------------------------------------------------------
__device__ __forceinline__ void stage_weights_lds(
    const _Float16* __restrict__ wg, _Float16* __restrict__ wl, int kd, int tid)
{
  const int chunks_per_col = kd / 8;           // 16B chunks per column
  const int total = 128 * chunks_per_col;
  for (int i = tid; i < total; i += 256) {
    int n = i / chunks_per_col;
    int c = i - n * chunks_per_col;
    *(uint4*)(wl + (size_t)n * (kd + 8) + c * 8) =
        *(const uint4*)(wg + (size_t)n * kd + c * 8);
  }
}

// ---------------------------------------------------------------------------
// Per-wave 16x128 GEMM strip on v_wmma_f32_16x16x32_f16.
// A and B both come from LDS (padded strides): fragment loads are
// ds_load_b128 pairs and the pre-WMMA wait is a short s_wait_dscnt instead of
// a full cache round-trip. K is compile time -> full unroll, imm offsets.
// ---------------------------------------------------------------------------
template <int KD>
__device__ __forceinline__ void wave_gemm_16x128(
    const _Float16* __restrict__ xin, int xstride,
    const _Float16* __restrict__ wlds, const float* __restrict__ bias,
    v8f (&acc)[8], int m_base, int lmod, int lhalf)
{
  constexpr int SW = KD + 8;                    // padded LDS weight stride
#pragma unroll
  for (int nt = 0; nt < 8; ++nt) {
    float b = bias[nt * 16 + lmod];
#pragma unroll
    for (int i = 0; i < 8; ++i) acc[nt][i] = b;
  }
  const _Float16* xrow = xin + (m_base + lmod) * xstride;
  const _Float16* wcol = wlds + (size_t)lmod * SW + lhalf * 16;
#pragma unroll
  for (int k = 0; k < KD; k += 32) {
    // A fragment (contiguous K, matching B): lanes<16 hold K=k..k+15,
    // lanes>=16 hold K=k+16..k+31.
    v8h lo = *(const v8h*)(xrow + k + lhalf * 16);
    v8h hi = *(const v8h*)(xrow + k + lhalf * 16 + 8);
    v16h afrag;
#pragma unroll
    for (int i = 0; i < 8; ++i) { afrag[i] = lo[i]; afrag[i + 8] = hi[i]; }
#pragma unroll
    for (int nt = 0; nt < 8; ++nt) {
      // B fragment: lane = column; lanes<16 carry K=k..k+15, lanes>=16 K=k+16..31.
      const v16h bfrag = *(const v16h*)(wcol + (size_t)nt * 16 * SW + k);
      acc[nt] = __builtin_amdgcn_wmma_f32_16x16x32_f16(
          false, afrag, false, bfrag, (short)0, acc[nt], false, false);
    }
  }
}

__device__ __forceinline__ void wave_store_act_relu(
    const v8f (&acc)[8], _Float16* __restrict__ act, int astride,
    int m_base, int lmod, int lhalf)
{
#pragma unroll
  for (int nt = 0; nt < 8; ++nt)
#pragma unroll
    for (int r = 0; r < 8; ++r) {
      float v = acc[nt][r];
      v = v > 0.0f ? v : 0.0f;                       // ReLU on hidden layers
      act[(m_base + r + lhalf * 8) * astride + nt * 16 + lmod] = (_Float16)v;
    }
}

__device__ __forceinline__ void wave_store_out(
    const v8f (&acc)[8], float* __restrict__ out, int row0, int nrows,
    int m_base, int lmod, int lhalf)
{
#pragma unroll
  for (int nt = 0; nt < 8; ++nt)
#pragma unroll
    for (int r = 0; r < 8; ++r) {
      int grow = row0 + m_base + r + lhalf * 8;
      if (grow < nrows)
        out[(size_t)grow * HID + nt * 16 + lmod] = acc[nt][r];
    }
}

// ---------------------------------------------------------------------------
// Fused 3-layer MLP: [x0|x1|x2] (NPARTS*128 wide fp32) -> 128 fp32.
// 128 rows per block, 8 waves, one 16-row strip per wave.
// LDS: X tile + act ping + all three weight matrices (padded); layer-3 input
// aliases the dead X region. flip_half>0 mirrors part-2 rows about flip_half.
// Totals: 174,080 / 239,616 / 305,152 B for NPARTS=1/2/3  (WGP has 320KB).
// ---------------------------------------------------------------------------
template <int NPARTS>
__global__ __launch_bounds__(256) void fused_mlp3_kernel(
    const float* __restrict__ x0, const float* __restrict__ x1,
    const float* __restrict__ x2, int nrows, int flip_half,
    const _Float16* __restrict__ w1, const float* __restrict__ b1,
    const _Float16* __restrict__ w2, const float* __restrict__ b2,
    const _Float16* __restrict__ w3, const float* __restrict__ b3,
    float* __restrict__ out)
{
  constexpr int K1 = NPARTS * HID;
  constexpr int SX = K1 + 8;   // +16B pad: rows stay 16B aligned, 4-bank stagger
  constexpr int SA = HID + 8;
  extern __shared__ _Float16 smem[];
  _Float16* X   = smem;                        // layer-1 input tile
  _Float16* A1  = X  + 128 * SX;               // layer-1 out / layer-2 in
  _Float16* W1l = A1 + 128 * SA;               // 128 x (K1+8)
  _Float16* W2l = W1l + 128 * SX;              // 128 x 136
  _Float16* W3l = W2l + 128 * SA;              // 128 x 136
  _Float16* A2  = X;                           // layer-2 out aliases X (dead)

  const int tid   = threadIdx.x;
  const int lane  = tid & 31;
  const int lmod  = lane & 15;
  const int lhalf = lane >> 4;
  const int m_base = (tid >> 5) * 16;
  const int row0   = blockIdx.x * 128;

  // Stage all weights into LDS (read once per block, reused by 8 waves).
  stage_weights_lds(w1, W1l, K1,  tid);
  stage_weights_lds(w2, W2l, HID, tid);
  stage_weights_lds(w3, W3l, HID, tid);

  // Stage concatenated fp32 input tile into LDS as f16 (zero-pad the tail).
  const float* parts[3] = { x0, x1, x2 };
  for (int idx = tid; idx < 128 * NPARTS * 32; idx += 256) {
    int r    = idx / (NPARTS * 32);
    int cq   = idx - r * (NPARTS * 32);
    int part = cq >> 5;
    int q    = cq & 31;
    int grow = row0 + r;
    v4h h = {};
    if (grow < nrows) {
      int srow = grow;
      if (NPARTS == 3 && part == 2 && flip_half > 0)
        srow = (grow < flip_half) ? grow + flip_half : grow - flip_half;
      const float4 f = ((const float4*)(parts[part] + (size_t)srow * HID))[q];
      h[0] = (_Float16)f.x; h[1] = (_Float16)f.y;
      h[2] = (_Float16)f.z; h[3] = (_Float16)f.w;
    }
    *(v4h*)(X + r * SX + part * HID + q * 4) = h;
  }
  __syncthreads();

  v8f acc[8];
  wave_gemm_16x128<K1>(X, SX, W1l, b1, acc, m_base, lmod, lhalf);    // layer 1
  wave_store_act_relu(acc, A1, SA, m_base, lmod, lhalf);
  __syncthreads();
  wave_gemm_16x128<HID>(A1, SA, W2l, b2, acc, m_base, lmod, lhalf);  // layer 2
  wave_store_act_relu(acc, A2, SA, m_base, lmod, lhalf);
  __syncthreads();
  wave_gemm_16x128<HID>(A2, SA, W3l, b3, acc, m_base, lmod, lhalf);  // layer 3
  wave_store_out(acc, out, row0, nrows, m_base, lmod, lhalf);
}

// ---------------------------------------------------------------------------
// Segment-sum scatter: one wave per edge, 512B row gather + fp32 HW atomics.
// out[widx[e]+w_off] += msg[ridx[e]+r_off]
// ---------------------------------------------------------------------------
__global__ __launch_bounds__(256) void scatter_add_kernel(
    const float* __restrict__ msg, int r_off,
    const int* __restrict__ ridx, const int* __restrict__ widx, int w_off,
    float* __restrict__ out, int n_edges)
{
  int e = blockIdx.x * 8 + (threadIdx.x >> 5);
  if (e >= n_edges) return;
  int lane = threadIdx.x & 31;
  const float4 v = ((const float4*)(msg + (size_t)(ridx[e] + r_off) * HID))[lane];
  float* d = out + (size_t)(widx[e] + w_off) * HID + lane * 4;
  unsafeAtomicAdd(d + 0, v.x);
  unsafeAtomicAdd(d + 1, v.y);
  unsafeAtomicAdd(d + 2, v.z);
  unsafeAtomicAdd(d + 3, v.w);
}

// ---------------------------------------------------------------------------
extern "C" void kernel_launch(void* const* d_in, const int* in_sizes, int n_in,
                              void* d_out, int out_size, void* d_ws, size_t ws_size,
                              hipStream_t stream)
{
  (void)in_sizes; (void)n_in; (void)out_size; (void)ws_size;

  const float* l_emb0  = (const float*)d_in[0];
  const float* c_emb0  = (const float*)d_in[1];
  const int*   pos_src = (const int*)d_in[2];
  const int*   pos_dst = (const int*)d_in[3];
  const int*   neg_src = (const int*)d_in[4];
  const int*   neg_dst = (const int*)d_in[5];
  // params flattened as W0,b0,W1,b1,W2,b2 per MLP, dict order:
  // l_msg_p(6..11), c_msg_p(12..17), l_upd_p(18..23), c_upd_p(24..29)
  const float* lmW[3] = {(const float*)d_in[6],  (const float*)d_in[8],  (const float*)d_in[10]};
  const float* lmB[3] = {(const float*)d_in[7],  (const float*)d_in[9],  (const float*)d_in[11]};
  const float* cmW[3] = {(const float*)d_in[12], (const float*)d_in[14], (const float*)d_in[16]};
  const float* cmB[3] = {(const float*)d_in[13], (const float*)d_in[15], (const float*)d_in[17]};
  const float* luW[3] = {(const float*)d_in[18], (const float*)d_in[20], (const float*)d_in[22]};
  const float* luB[3] = {(const float*)d_in[19], (const float*)d_in[21], (const float*)d_in[23]};
  const float* cuW[3] = {(const float*)d_in[24], (const float*)d_in[26], (const float*)d_in[28]};
  const float* cuB[3] = {(const float*)d_in[25], (const float*)d_in[27], (const float*)d_in[29]};

  // ---- workspace carve ----
  const size_t LB = (size_t)NLITS * HID * sizeof(float);   // 102.4 MB
  const size_t CB = (size_t)NCLS  * HID * sizeof(float);   // 204.8 MB
  char* w = (char*)d_ws;
  auto carve = [&](size_t bytes) {
    char* p = w; w += (bytes + 255) & ~(size_t)255; return p;
  };
  float* lA    = (float*)carve(LB);
  float* lBuf  = (float*)carve(LB);
  float* l_msg = (float*)carve(LB);
  float* c2l   = (float*)carve(LB);
  float* cA    = (float*)carve(CB);
  float* cBuf  = (float*)carve(CB);
  float* c_msg = (float*)carve(CB);
  float* l2c   = (float*)carve(CB);

  // f16 transposed weights
  _Float16* wtl[3]; _Float16* wtc[3]; _Float16* wtu[3]; _Float16* wtq[3];
  for (int i = 0; i < 3; ++i) wtl[i] = (_Float16*)carve((size_t)128 * HID * 2);
  for (int i = 0; i < 3; ++i) wtc[i] = (_Float16*)carve((size_t)128 * HID * 2);
  wtu[0] = (_Float16*)carve((size_t)384 * HID * 2);
  wtu[1] = (_Float16*)carve((size_t)128 * HID * 2);
  wtu[2] = (_Float16*)carve((size_t)128 * HID * 2);
  wtq[0] = (_Float16*)carve((size_t)256 * HID * 2);
  wtq[1] = (_Float16*)carve((size_t)128 * HID * 2);
  wtq[2] = (_Float16*)carve((size_t)128 * HID * 2);

  // ---- weight transpose+convert (cheap, deterministic every call) ----
  auto wprep = [&](const float* W, _Float16* Wt, int K) {
    int elems = K * HID;
    wprep_kernel<<<(elems + 255) / 256, 256, 0, stream>>>(W, Wt, K, HID);
  };
  for (int i = 0; i < 3; ++i) { wprep(lmW[i], wtl[i], 128); wprep(cmW[i], wtc[i], 128); }
  wprep(luW[0], wtu[0], 384); wprep(luW[1], wtu[1], 128); wprep(luW[2], wtu[2], 128);
  wprep(cuW[0], wtq[0], 256); wprep(cuW[1], wtq[1], 128); wprep(cuW[2], wtq[2], 128);

  // dynamic LDS sizes: X + act + 3 padded weight arrays (WGP has 320KB)
  const int SAB = 128 * (HID + 8) * 2;                 // 34,816 B (act / 128-K W)
  const int lds1 = 128 * (1 * HID + 8) * 2 * 2 + 3 * SAB;   // 174,080 B
  const int lds2 = 128 * (2 * HID + 8) * 2 * 2 + 3 * SAB;   // 239,616 B
  const int lds3 = 128 * (3 * HID + 8) * 2 * 2 + 3 * SAB;   // 305,152 B
  (void)hipFuncSetAttribute(reinterpret_cast<const void*>(&fused_mlp3_kernel<1>),
                            hipFuncAttributeMaxDynamicSharedMemorySize, lds1);
  (void)hipFuncSetAttribute(reinterpret_cast<const void*>(&fused_mlp3_kernel<2>),
                            hipFuncAttributeMaxDynamicSharedMemorySize, lds2);
  (void)hipFuncSetAttribute(reinterpret_cast<const void*>(&fused_mlp3_kernel<3>),
                            hipFuncAttributeMaxDynamicSharedMemorySize, lds3);

  hipMemcpyAsync(lA, l_emb0, LB, hipMemcpyDeviceToDevice, stream);
  hipMemcpyAsync(cA, c_emb0, CB, hipMemcpyDeviceToDevice, stream);

  const dim3 blk(256);
  const dim3 gridL((NLITS + 127) / 128);
  const dim3 gridC((NCLS  + 127) / 128);
  const dim3 gridE((NEDGES + 7) / 8);

  float* l_cur = lA;   float* c_cur = cA;
  float* l_nxt = lBuf; float* c_nxt = cBuf;

  for (int r = 0; r < NROUND; ++r) {
    // literal / clause message MLPs (fused 3 layers, WMMA)
    fused_mlp3_kernel<1><<<gridL, blk, lds1, stream>>>(
        l_cur, nullptr, nullptr, NLITS, 0,
        wtl[0], lmB[0], wtl[1], lmB[1], wtl[2], lmB[2], l_msg);
    fused_mlp3_kernel<1><<<gridC, blk, lds1, stream>>>(
        c_cur, nullptr, nullptr, NCLS, 0,
        wtc[0], cmB[0], wtc[1], cmB[1], wtc[2], cmB[2], c_msg);

    // segment sums
    hipMemsetAsync(l2c, 0, CB, stream);
    hipMemsetAsync(c2l, 0, LB, stream);
    scatter_add_kernel<<<gridE, blk, 0, stream>>>(l_msg, 0,     pos_src, pos_dst, 0,     l2c, NEDGES);
    scatter_add_kernel<<<gridE, blk, 0, stream>>>(l_msg, NVARS, neg_src, neg_dst, 0,     l2c, NEDGES);
    scatter_add_kernel<<<gridE, blk, 0, stream>>>(c_msg, 0,     pos_dst, pos_src, 0,     c2l, NEDGES);
    scatter_add_kernel<<<gridE, blk, 0, stream>>>(c_msg, 0,     neg_dst, neg_src, NVARS, c2l, NEDGES);

    // update MLPs; last round writes straight into d_out (l_emb || c_emb)
    float* l_out = (r == NROUND - 1) ? (float*)d_out : l_nxt;
    float* c_out = (r == NROUND - 1) ? (float*)d_out + (size_t)NLITS * HID : c_nxt;
    fused_mlp3_kernel<2><<<gridC, blk, lds2, stream>>>(
        l2c, c_cur, nullptr, NCLS, 0,
        wtq[0], cuB[0], wtq[1], cuB[1], wtq[2], cuB[2], c_out);
    fused_mlp3_kernel<3><<<gridL, blk, lds3, stream>>>(
        c2l, l_cur, l_cur, NLITS, NVARS,
        wtu[0], luB[0], wtu[1], luB[1], wtu[2], luB[2], l_out);

    float* t;
    t = l_cur; l_cur = l_nxt; l_nxt = t;
    t = c_cur; c_cur = c_nxt; c_nxt = t;
  }
}
